// Model_50869592654858
// MI455X (gfx1250) — compile-verified, matrix operand order checked by reading, not run on hardware
//
#include <hip/hip_runtime.h>
#include <hip/hip_bf16.h>

// ---------------------------------------------------------------------------
// MI455X (gfx1250) bidirectional 3-layer LSTM over a packed sequence.
//
// Strategy (see analysis): each workgroup owns 16 batch rows (independent
// sequences) and performs the whole 128-step scan for one (layer,direction).
// Per step it computes gates(16x1600) = [x_t | h](16xKc) @ Wc^T with
// v_wmma_f32_16x16x32_bf16, weights pre-packed into the exact B-fragment
// lane layout (streamed from L2: 23MB packed << 192MB L2). Cell state c is
// held in f32 VGPRs; h is recirculated as bf16 through LDS.
// ---------------------------------------------------------------------------

typedef __attribute__((ext_vector_type(16))) __bf16 v16bf;
typedef __attribute__((ext_vector_type(8)))  float  v8f;

#define HID    400
#define GATES  1600
#define TMAX   128
#define BMAX   1024
#define KCMAX  1216      // max (din + HID) padded to mult of 32
#define NTILES 100       // 1600 / 16
#define KTMAX  38        // ceil(1200/32)
#define NLC    6         // layer*2 + dir combos

// workspace layout (bytes)
static constexpr size_t OFF_OFFS = 0;                          // 128 * i32
static constexpr size_t OFF_BIAS = 1024;                       // 6*1600 f32
static constexpr size_t OFF_WPK  = 64ull * 1024;               // packed bf16 weights
static constexpr size_t OFF_BUFX = 24ull  * 1024 * 1024;       // dense x, 131072x400 bf16
static constexpr size_t OFF_BUFA = 132ull * 1024 * 1024;       // 131072x800 bf16
static constexpr size_t OFF_BUFB = 336ull * 1024 * 1024;       // 131072x800 bf16
static constexpr size_t BUFAB_BYTES = (size_t)TMAX * BMAX * 800 * 2;
static constexpr size_t ZERO_BYTES  = (OFF_BUFB + BUFAB_BYTES) - OFF_BUFX; // 512MB

// ---------------------------------------------------------------------------
__global__ void zero_ws_kernel(uint4* __restrict__ p, long n) {
  long i = (long)blockIdx.x * blockDim.x + threadIdx.x;
  long stride = (long)gridDim.x * blockDim.x;
  uint4 z; z.x = z.y = z.z = z.w = 0u;
  for (; i < n; i += stride) p[i] = z;
}

__global__ void prep_offs_kernel(const int* __restrict__ bs, int* __restrict__ offs) {
  if (blockIdx.x == 0 && threadIdx.x == 0) {
    int acc = 0;
    for (int t = 0; t < TMAX; ++t) { offs[t] = acc; acc += bs[t]; }
  }
}

// packed x (f32) -> dense (t, b, 0..399) bf16; invalid entries stay zero
__global__ void pack_x_kernel(const float* __restrict__ x, const int* __restrict__ bs,
                              const int* __restrict__ offs, __bf16* __restrict__ bufX) {
  int b = blockIdx.x, t = blockIdx.y;
  if (b >= bs[t]) return;
  size_t src = (size_t)(offs[t] + b) * 400;
  size_t dst = ((size_t)t * BMAX + b) * 400;
  for (int j = threadIdx.x; j < 400; j += blockDim.x)
    bufX[dst + j] = (__bf16)x[src + j];
}

// pack Wc = [Wih | Whh] (1600 x Kc) into WMMA B-fragment order:
// wpk[(((lc*100+nt)*38+kt)*32+lane)*16 + e] = Wc[nt*16+(lane&15)][kt*32+kidx(e,lane)]
__global__ void pack_w_kernel(const float* __restrict__ Wih, const float* __restrict__ Whh,
                              __bf16* __restrict__ wpk) {
  long gid = (long)blockIdx.x * blockDim.x + threadIdx.x;
  const long total = (long)NLC * NTILES * KTMAX * 32;
  if (gid >= total) return;
  int lane = (int)(gid & 31); long r = gid >> 5;
  int kt = (int)(r % KTMAX); r /= KTMAX;
  int nt = (int)(r % NTILES); int lc = (int)(r / NTILES);
  int din = (lc < 2) ? 400 : 800;
  int n   = nt * 16 + (lane & 15);
  int khi = ((lane >> 4) & 1) << 3;
  v16bf ov;
#pragma unroll
  for (int e = 0; e < 16; ++e) {
    int k = kt * 32 + (((e >> 1) & 3) << 1) + (e & 1) + ((e >> 3) << 4) + khi;
    float v = 0.f;
    if (k < din)            v = Wih[((size_t)lc * GATES + n) * 800 + k];
    else if (k < din + HID) v = Whh[((size_t)lc * GATES + n) * HID + (k - din)];
    ov[e] = (__bf16)v;
  }
  *(v16bf*)&wpk[(size_t)gid * 16] = ov;
}

__global__ void pack_bias_kernel(const float* __restrict__ bih, const float* __restrict__ bhh,
                                 float* __restrict__ bias) {
  int gid = blockIdx.x * blockDim.x + threadIdx.x;
  if (gid < NLC * GATES) bias[gid] = bih[gid] + bhh[gid];
}

// ---------------------------------------------------------------------------
// Recurrent scan: one block = 16 batch rows, whole time loop for one (lc,dir).
__global__ __launch_bounds__(256) void lstm_scan_kernel(
    const __bf16* __restrict__ in_buf, int din,
    __bf16* __restrict__ out_buf,
    const __bf16* __restrict__ wpk,
    const float* __restrict__ bias,
    const int* __restrict__ bs,
    float* __restrict__ hn, float* __restrict__ cn,
    int lc, int dir)
{
  __shared__ __bf16 xh[16 * KCMAX];   // A operand: [x_t | h | zero-pad], bf16
  __shared__ __bf16 h2[16 * HID];     // double-buffered new h

  const int tid     = threadIdx.x;
  const int lane    = tid & 31;
  const int wave    = tid >> 5;
  const int rowbase = blockIdx.x * 16;
  const int m       = lane & 15;
  const int khi     = ((lane >> 4) & 1) << 3;
  const int KT      = (din + HID + 31) >> 5;    // 25 or 38

  // each wave owns up to 4 column-quads; quad q -> gate cols [q*16, q*16+16)
  int quads[4]; int nq = 0;
  for (int q = wave; q < 25; q += 8) quads[nq++] = q;
  nq = __builtin_amdgcn_readfirstlane(nq);      // force uniform branch around WMMA

  // hoist combined biases (i,f,g,o) per quad
  float bsv[4][4];
#pragma unroll
  for (int qi = 0; qi < 4; ++qi) {
    int q = quads[qi < nq ? qi : 0];
    int col = q * 16 + m;
#pragma unroll
    for (int g = 0; g < 4; ++g)
      bsv[qi][g] = bias[lc * GATES + g * HID + col];
  }

  // cell state in f32 VGPRs
  float creg[4][8];
#pragma unroll
  for (int qi = 0; qi < 4; ++qi)
#pragma unroll
    for (int j = 0; j < 8; ++j) creg[qi][j] = 0.f;

  // zero h region + K padding of the staged A operand
  for (int i = tid; i < 16 * (KCMAX - din); i += 256) {
    int r = i / (KCMAX - din), c = i % (KCMAX - din);
    xh[r * KCMAX + din + c] = (__bf16)0.f;
  }

  const int dchunks = din >> 2;   // 4 bf16 per copy chunk
  for (int tt = 0; tt < TMAX; ++tt) {
    const int t = dir ? (TMAX - 1 - tt) : tt;
    const int bsT = bs[t];
    if (rowbase >= bsT) { if (dir) continue; else break; }  // uniform per block

    // stage x_t into A cols [0, din)
    for (int i = tid; i < 16 * dchunks; i += 256) {
      int r = i / dchunks, k4 = (i - r * dchunks) << 2;
      *(uint2*)&xh[r * KCMAX + k4] =
          *(const uint2*)&in_buf[((size_t)(t * BMAX) + rowbase + r) * din + k4];
    }
    __syncthreads();

#pragma unroll
    for (int qi = 0; qi < 4; ++qi) {
      if (qi >= nq) continue;                   // wave-uniform
      const int q   = quads[qi];
      const int col = q * 16 + m;
      v8f acc[4] = { v8f{}, v8f{}, v8f{}, v8f{} };

      for (int kt = 0; kt < KT; ++kt) {
        // A fragment from LDS (16-bit A layout, pairs are b32-contiguous)
        union { v16bf v; unsigned int u[8]; } A;
        const int kb = kt << 5;
#pragma unroll
        for (int p = 0; p < 8; ++p) {
          int k = kb + ((p & 3) << 1) + ((p >> 2) << 4) + khi;
          A.u[p] = *(const unsigned int*)&xh[m * KCMAX + k];
        }
#pragma unroll
        for (int g = 0; g < 4; ++g) {
          int nt = g * 25 + q;
          size_t widx = ((((size_t)lc * NTILES + nt) * KTMAX + kt) * 32 + lane) * 16;
          const v16bf B = *(const v16bf*)&wpk[widx];   // 32B -> 2x global_load_b128
          acc[g] = __builtin_amdgcn_wmma_f32_16x16x32_bf16(
              false, A.v, false, B, (short)0, acc[g], false, false);
        }
      }

      // LSTM cell entirely in registers (D layout: row = j + 8*(lane>>4))
#pragma unroll
      for (int j = 0; j < 8; ++j) {
        const int row  = j + ((lane >> 4) << 3);
        const int grow = rowbase + row;
        float xi = acc[0][j] + bsv[qi][0];
        float xf = acc[1][j] + bsv[qi][1];
        float xg = acc[2][j] + bsv[qi][2];
        float xo = acc[3][j] + bsv[qi][3];
        float ig = 1.f / (1.f + __expf(-xi));
        float fg = 1.f / (1.f + __expf(-xf));
        float og = 1.f / (1.f + __expf(-xo));
        float e2 = __expf(2.f * xg);
        float gg = (e2 - 1.f) / (e2 + 1.f);
        float cnew = fg * creg[qi][j] + ig * gg;
        float ec = __expf(2.f * cnew);
        float tc = (ec - 1.f) / (ec + 1.f);
        float hnew = og * tc;
        bool valid = grow < bsT;
        if (valid) creg[qi][j] = cnew;
        float hheld = valid ? hnew : (float)xh[row * KCMAX + din + col];
        h2[row * HID + col] = (__bf16)hheld;
        out_buf[((size_t)(t * BMAX) + grow) * 800 + dir * HID + col] =
            (__bf16)(valid ? hnew : 0.f);
      }
    }
    __syncthreads();
    // commit held h for the next step
    for (int i = tid; i < 16 * (HID >> 2); i += 256) {
      int r = i / (HID >> 2), c4 = (i - r * (HID >> 2)) << 2;
      *(uint2*)&xh[r * KCMAX + din + c4] = *(const uint2*)&h2[r * HID + c4];
    }
  }
  __syncthreads();

  // final states h_n (bf16-held -> f32) and c_n (full f32 from VGPRs)
  for (int i = tid; i < 16 * HID; i += 256) {
    int r = i / HID, c = i - r * HID;
    hn[((size_t)lc * BMAX + rowbase + r) * HID + c] = (float)xh[r * KCMAX + din + c];
  }
#pragma unroll
  for (int qi = 0; qi < 4; ++qi) {
    if (qi >= nq) continue;
    const int col = quads[qi] * 16 + m;
#pragma unroll
    for (int j = 0; j < 8; ++j) {
      const int row = j + ((lane >> 4) << 3);
      cn[((size_t)lc * BMAX + rowbase + row) * HID + col] = creg[qi][j];
    }
  }
}

// dense last-layer output (bf16) -> packed f32
__global__ void gather_out_kernel(const __bf16* __restrict__ buf, const int* __restrict__ bs,
                                  const int* __restrict__ offs, float* __restrict__ out) {
  int b = blockIdx.x, t = blockIdx.y;
  if (b >= bs[t]) return;
  size_t dst = (size_t)(offs[t] + b) * 800;
  size_t src = ((size_t)t * BMAX + b) * 800;
  for (int j = threadIdx.x; j < 800; j += blockDim.x)
    out[dst + j] = (float)buf[src + j];
}

// ---------------------------------------------------------------------------
extern "C" void kernel_launch(void* const* d_in, const int* in_sizes, int n_in,
                              void* d_out, int out_size, void* d_ws, size_t ws_size,
                              hipStream_t stream) {
  (void)in_sizes; (void)n_in; (void)out_size; (void)ws_size;
  const float* x   = (const float*)d_in[0];
  const float* Wih = (const float*)d_in[1];
  const float* Whh = (const float*)d_in[2];
  const float* bih = (const float*)d_in[3];
  const float* bhh = (const float*)d_in[4];
  const int*   bsv = (const int*)d_in[5];

  char* ws = (char*)d_ws;
  int*    offs = (int*)(ws + OFF_OFFS);
  float*  bias = (float*)(ws + OFF_BIAS);
  __bf16* wpk  = (__bf16*)(ws + OFF_WPK);
  __bf16* bufX = (__bf16*)(ws + OFF_BUFX);
  __bf16* bufA = (__bf16*)(ws + OFF_BUFA);
  __bf16* bufB = (__bf16*)(ws + OFF_BUFB);

  float* out_packed = (float*)d_out;                        // 65536*800
  float* hn = out_packed + (size_t)65536 * 800;             // 6*1024*400
  float* cn = hn + (size_t)NLC * BMAX * HID;                // 6*1024*400

  // 1) zero activation buffers (skipped (t,row) regions must read as 0)
  zero_ws_kernel<<<4096, 256, 0, stream>>>((uint4*)(ws + OFF_BUFX), (long)(ZERO_BYTES / 16));
  // 2) offsets, input pack, weight/bias pack
  prep_offs_kernel<<<1, 32, 0, stream>>>(bsv, offs);
  pack_x_kernel<<<dim3(BMAX, TMAX), 128, 0, stream>>>(x, bsv, offs, bufX);
  {
    long total = (long)NLC * NTILES * KTMAX * 32;
    pack_w_kernel<<<(unsigned)((total + 255) / 256), 256, 0, stream>>>(Wih, Whh, wpk);
  }
  pack_bias_kernel<<<(NLC * GATES + 255) / 256, 256, 0, stream>>>(bih, bhh, bias);

  // 3) scans: L0 bufX->bufA, L1 bufA->bufB, L2 bufB->bufA (64 blocks = 1024/16 rows)
  lstm_scan_kernel<<<64, 256, 0, stream>>>(bufX, 400, bufA, wpk, bias, bsv, hn, cn, 0, 0);
  lstm_scan_kernel<<<64, 256, 0, stream>>>(bufX, 400, bufA, wpk, bias, bsv, hn, cn, 1, 1);
  lstm_scan_kernel<<<64, 256, 0, stream>>>(bufA, 800, bufB, wpk, bias, bsv, hn, cn, 2, 0);
  lstm_scan_kernel<<<64, 256, 0, stream>>>(bufA, 800, bufB, wpk, bias, bsv, hn, cn, 3, 1);
  lstm_scan_kernel<<<64, 256, 0, stream>>>(bufB, 800, bufA, wpk, bias, bsv, hn, cn, 4, 0);
  lstm_scan_kernel<<<64, 256, 0, stream>>>(bufB, 800, bufA, wpk, bias, bsv, hn, cn, 5, 1);

  // 4) pack last-layer output to f32
  gather_out_kernel<<<dim3(BMAX, TMAX), 256, 0, stream>>>(bufA, bsv, offs, out_packed);
}
